// W4A4Linear_14843406975731
// MI455X (gfx1250) — 8-bit matrix operand order repaired and checked by reading, compiled, not run on hardware
//
#include <hip/hip_runtime.h>
#include <hip/hip_bf16.h>
#include <stdint.h>

#define D_IN   4096
#define D_OUT  11008
#define M_TOK  8192
#define KB128  (D_IN / 128)
#define M16    (M_TOK / 16)
#define N16    (D_OUT / 16)

typedef __attribute__((ext_vector_type(16))) int   v16i;
typedef __attribute__((ext_vector_type(8)))  float v8f;

__device__ __forceinline__ unsigned int f32_to_e4m3(float f) {
    unsigned int u = __float_as_uint(f);
    unsigned int s = (u >> 24) & 0x80u;
    float a = __uint_as_float(u & 0x7FFFFFFFu);
    if (a >= 464.0f) return s | 0x7Eu;
    if (a < 0.000976562f) return s;
    int exp = (int)((u >> 23) & 0xFFu) - 127;
    if (exp < -6) {
        float q = rintf(a * 512.0f);
        unsigned int mi = (unsigned int)q;
        if (mi == 0u) return s;
        if (mi >= 8u) return s | 0x08u;
        return s | mi;
    }
    float q = rintf(ldexpf(a, 3 - exp));
    unsigned int mi = (unsigned int)q;
    if (mi == 16u) { mi = 8u; exp += 1; if (exp > 8) return s | 0x7Eu; }
    return s | (unsigned int)((exp + 7) << 3) | (mi - 8u);
}

__device__ __forceinline__ float e4m3_to_f32(unsigned int b) {
    unsigned int s = (b >> 7) & 1u;
    int e = (int)((b >> 3) & 15u);
    int m = (int)(b & 7u);
    float v = (e == 0) ? ldexpf((float)m, -9)
                       : ldexpf((float)(8 + m), e - 10);
    return s ? -v : v;
}

__global__ void __launch_bounds__(256)
quant_act_kernel(const float* __restrict__ x,
                 unsigned char* __restrict__ A8,
                 float* __restrict__ scales)
{
    __shared__ float red[256];
    const int r   = blockIdx.x;
    const int tid = threadIdx.x;

    const float4* xr = (const float4*)(x + (size_t)r * D_IN);
    float vals[16];
    float4 t0 = xr[tid * 4 + 0];
    float4 t1 = xr[tid * 4 + 1];
    float4 t2 = xr[tid * 4 + 2];
    float4 t3 = xr[tid * 4 + 3];
    vals[0]=t0.x;  vals[1]=t0.y;  vals[2]=t0.z;  vals[3]=t0.w;
    vals[4]=t1.x;  vals[5]=t1.y;  vals[6]=t1.z;  vals[7]=t1.w;
    vals[8]=t2.x;  vals[9]=t2.y;  vals[10]=t2.z; vals[11]=t2.w;
    vals[12]=t3.x; vals[13]=t3.y; vals[14]=t3.z; vals[15]=t3.w;

    float am = 0.0f;
#pragma unroll
    for (int i = 0; i < 16; ++i) am = fmaxf(am, fabsf(vals[i]));
    red[tid] = am;
    __syncthreads();
#pragma unroll
    for (int s = 128; s > 0; s >>= 1) {
        if (tid < s) red[tid] = fmaxf(red[tid], red[tid + s]);
        __syncthreads();
    }
    const float scale = fmaxf(red[0], 1e-5f) * (1.0f / 7.0f);
    const float inv   = 1.0f / scale;
    if (tid == 0) scales[r] = scale;

    const int k0  = tid * 16;
    const int kb  = k0 >> 7;
    const int d0  = (k0 >> 2) & 31;
    const int vst = (d0 >> 3) << 2;
    unsigned int dw[4] = {0u, 0u, 0u, 0u};
#pragma unroll
    for (int i = 0; i < 16; ++i) {
        unsigned int b = f32_to_e4m3(rintf(vals[i] * inv));
        dw[i >> 2] |= b << (8 * (i & 3));
    }
    unsigned int* A32 = (unsigned int*)A8;
    const size_t baseDW = ((size_t)((r >> 4) * KB128 + kb) << 9);
    const int lane0 = r & 15;
    *(uint4*)(A32 + baseDW + (size_t)(lane0 + (((d0 >> 2) & 1) << 4)) * 16 + vst) = make_uint4(dw[0], dw[1], dw[2], dw[3]);
}

__global__ void __launch_bounds__(256)
quant_w_kernel(const float* __restrict__ W,
               unsigned char* __restrict__ Whi,
               unsigned char* __restrict__ Wlo,
               int use_lo)
{
    const int n   = blockIdx.x;
    const int tid = threadIdx.x;
    const int k0  = tid * 16;
    const float* wr = W + (size_t)n * D_IN + k0;

    const int kb = k0 >> 7;
    const int d0 = (k0 >> 2) & 31;
    const int v0 = (d0 >> 3) * 4;
    const int L  = (n & 15) + (((d0 >> 2) & 1) << 4);
    const size_t base = ((size_t)((n >> 4) * KB128 + kb) << 11) + (size_t)L * 64 + (size_t)v0 * 4;

    unsigned int hi[4] = {0u,0u,0u,0u};
    unsigned int lo[4] = {0u,0u,0u,0u};
#pragma unroll
    for (int j = 0; j < 4; ++j) {
#pragma unroll
        for (int b = 0; b < 4; ++b) {
            float w = wr[j * 4 + b];
            unsigned int h8 = f32_to_e4m3(w);
            hi[j] |= h8 << (8 * b);
            float rres = w - e4m3_to_f32(h8);
            lo[j] |= f32_to_e4m3(rres) << (8 * b);
        }
    }
    *(uint4*)(Whi + base) = make_uint4(hi[0], hi[1], hi[2], hi[3]);
    if (use_lo) *(uint4*)(Wlo + base) = make_uint4(lo[0], lo[1], lo[2], lo[3]);
}

#define WMMA_FP8(C, A, B) \
    (C) = __builtin_amdgcn_wmma_f32_16x16x128_fp8_fp8((A), (B), (short)0, (C), false, false)

__device__ __forceinline__ void epilogue_tile(float* __restrict__ out,
                                              const float* __restrict__ scales,
                                              const float* __restrict__ bias,
                                              v8f c, int mB, int n)
{
    const float bv = bias[n];
#pragma unroll
    for (int j = 0; j < 8; ++j)
        out[(size_t)(mB + j) * D_OUT + n] = c[j] * scales[mB + j] + bv;
}

template <int USE_LO>
__global__ void __launch_bounds__(256)
gemm_fp8_kernel(const unsigned char* __restrict__ Asw,
                const unsigned char* __restrict__ Whi,
                const unsigned char* __restrict__ Wlo,
                const float* __restrict__ scales,
                const float* __restrict__ bias,
                float* __restrict__ out)
{
    const int lane = threadIdx.x & 31;
    const int wave = threadIdx.x >> 5;
    const int wm = wave & 1;
    const int wn = wave >> 1;
    const int M0 = blockIdx.x * 128 + wm * 64;
    const int N0 = blockIdx.y * 128 + wn * 32;

    const size_t laneOff = (size_t)lane * 64;

    const unsigned char* pa0 = Asw + (((size_t)((M0 >> 4) + 0) * KB128) << 11) + laneOff;
    const unsigned char* pa1 = Asw + (((size_t)((M0 >> 4) + 1) * KB128) << 11) + laneOff;
    const unsigned char* pa2 = Asw + (((size_t)((M0 >> 4) + 2) * KB128) << 11) + laneOff;
    const unsigned char* pa3 = Asw + (((size_t)((M0 >> 4) + 3) * KB128) << 11) + laneOff;
    const size_t bBase0 = (((size_t)((N0 >> 4) + 0) * KB128) << 11) + laneOff;
    const size_t bBase1 = (((size_t)((N0 >> 4) + 1) * KB128) << 11) + laneOff;

    v8f c00 = {}, c01 = {};
    v8f c10 = {}, c11 = {};
    v8f c20 = {}, c21 = {};
    v8f c30 = {}, c31 = {};

    for (int kb = 0; kb < KB128; ++kb) {
        const size_t ko = (size_t)kb << 11;

        const v16i a0 = *(const v16i*)(pa0 + ko);
        const v16i a1 = *(const v16i*)(pa1 + ko);
        const v16i a2 = *(const v16i*)(pa2 + ko);
        const v16i a3 = *(const v16i*)(pa3 + ko);
        const v16i b0 = *(const v16i*)(Whi + bBase0 + ko);
        const v16i b1 = *(const v16i*)(Whi + bBase1 + ko);

        if (kb + 1 < KB128) {
            __builtin_prefetch(pa0 + ko + 2048, 0, 1);
            __builtin_prefetch(Whi + bBase0 + ko + 2048, 0, 1);
        }

        WMMA_FP8(c00, a0, b0);
        WMMA_FP8(c01, a0, b1);
        WMMA_FP8(c10, a1, b0);
        WMMA_FP8(c11, a1, b1);
        WMMA_FP8(c20, a2, b0);
        WMMA_FP8(c21, a2, b1);
        WMMA_FP8(c30, a3, b0);
        WMMA_FP8(c31, a3, b1);

        if (USE_LO) {
            const v16i l0 = *(const v16i*)(Wlo + bBase0 + ko);
            const v16i l1 = *(const v16i*)(Wlo + bBase1 + ko);
            WMMA_FP8(c00, a0, l0);
            WMMA_FP8(c01, a0, l1);
            WMMA_FP8(c10, a1, l0);
            WMMA_FP8(c11, a1, l1);
            WMMA_FP8(c20, a2, l0);
            WMMA_FP8(c21, a2, l1);
            WMMA_FP8(c30, a3, l0);
            WMMA_FP8(c31, a3, l1);
        }
    }

    const int h   = lane >> 4;
    const int col = lane & 15;
    epilogue_tile(out, scales, bias, c00, M0 +  0 + h * 8, N0 +  0 + col);
    epilogue_tile(out, scales, bias, c01, M0 +  0 + h * 8, N0 + 16 + col);
    epilogue_tile(out, scales, bias, c10, M0 + 16 + h * 8, N0 +  0 + col);
    epilogue_tile(out, scales, bias, c11, M0 + 16 + h * 8, N0 + 16 + col);
    epilogue_tile(out, scales, bias, c20, M0 + 32 + h * 8, N0 +  0 + col);
    epilogue_tile(out, scales, bias, c21, M0 + 32 + h * 8, N0 + 16 + col);
    epilogue_tile(out, scales, bias, c30, M0 + 48 + h * 8, N0 +  0 + col);
    epilogue_tile(out, scales, bias, c31, M0 + 48 + h * 8, N0 + 16 + col);
}

extern "C" void kernel_launch(void* const* d_in, const int* in_sizes, int n_in,
                              void* d_out, int out_size, void* d_ws, size_t ws_size,
                              hipStream_t stream)
{
    const float* x    = (const float*)d_in[0];
    const float* W    = (const float*)d_in[1];
    const float* bias = (const float*)d_in[2];
    float* out        = (float*)d_out;

    constexpr size_t A8_BYTES = (size_t)M_TOK * D_IN;
    constexpr size_t SC_BYTES = (size_t)M_TOK * sizeof(float);
    constexpr size_t W8_BYTES = (size_t)D_OUT * D_IN;

    unsigned char* ws  = (unsigned char*)d_ws;
    unsigned char* A8  = ws;
    float*         sc  = (float*)(ws + A8_BYTES);
    unsigned char* Whi = ws + A8_BYTES + SC_BYTES;
    unsigned char* Wlo = Whi + W8_BYTES;

    const int use_lo = (ws_size >= A8_BYTES + SC_BYTES + 2 * W8_BYTES) ? 1 : 0;

    quant_w_kernel<<<D_OUT, 256, 0, stream>>>(W, Whi, Wlo, use_lo);
    quant_act_kernel<<<M_TOK, 256, 0, stream>>>(x, A8, sc);

    const dim3 grid(M_TOK / 128, D_OUT / 128);
    if (use_lo)
        gemm_fp8_kernel<1><<<grid, 256, 0, stream>>>(A8, Whi, Wlo, sc, bias, out);
    else
        gemm_fp8_kernel<0><<<grid, 256, 0, stream>>>(A8, Whi, Wlo, sc, bias, out);
}
